// KernelDensityEstimator_33423435497747
// MI455X (gfx1250) — compile-verified, matrix operand order checked by reading, not run on hardware
//
#include <hip/hip_runtime.h>

// KDE: out[i] = coef/N * sum_j exp(-(x[i]-t[j])^2 / (2 var))
//   Rewritten: e = exp2(-(d')^2), d' = s*x[i] - s*t[j], s = sqrt(log2(e)/(2 var))
// Compute-bound on v_exp_f32 (134M pair evals; inputs are only 96KB).
// WMMA carries the row-sum: D += A(16x32 f16 kernel vals) x B(ones).
// Train set streamed HBM->LDS with global_load_async_to_lds_b128, double-buffered.

typedef __attribute__((ext_vector_type(16))) _Float16 v16h;
typedef __attribute__((ext_vector_type(2)))  _Float16 v2h;
typedef __attribute__((ext_vector_type(2)))  __fp16   v2fp16;
typedef __attribute__((ext_vector_type(8)))  float    v8f;

#define BLOCK 256
#define WAVES (BLOCK / 32)
#define ROWS_PER_WAVE 16
#define ROWS_PER_BLOCK (WAVES * ROWS_PER_WAVE)   // 128 test rows per block
#define CHUNK 4096                               // floats per LDS buffer (16KB), x2 buffers

__device__ __forceinline__ void async_copy_chunk(float* lds, const float* gsrc,
                                                 int nfloats, int tid)
{
    // Low 32 bits of a generic pointer to LDS == LDS byte address (ISA 10.2).
    const uint32_t ldsBase = (uint32_t)(uintptr_t)lds;
    const uint64_t sbase   = (uint64_t)(uintptr_t)gsrc;
    const int n4 = nfloats >> 2;                 // whole float4s
    for (int i = tid; i < n4; i += BLOCK) {
        const uint32_t off   = (uint32_t)i << 4; // byte offset
        const uint32_t laddr = ldsBase + off;
        // GVS mode: vdst = LDS byte addr, vaddr = 32-bit byte offset, saddr = base
        asm volatile("global_load_async_to_lds_b128 %0, %1, %2"
                     :: "v"(laddr), "v"(off), "s"(sbase)
                     : "memory");
    }
}

__global__ __launch_bounds__(BLOCK)
void kde_wmma_kernel(const float* __restrict__ x,
                     const float* __restrict__ tr,
                     float* __restrict__ out,
                     int n_test, int n_train,
                     float sScale,      //  sqrt(log2(e)/(2 var))
                     float negS,        // -sScale
                     float outScale)    //  coef / n_train
{
    __shared__ __align__(16) float smem[2][CHUNK];

    const int tid     = threadIdx.x;
    const int lane    = tid & 31;
    const int wave    = tid >> 5;
    const int rowBase = blockIdx.x * ROWS_PER_BLOCK + wave * ROWS_PER_WAVE;

    // A-matrix row for this lane (rows striped across lanes 0..15 / 16..31)
    const int myRow = rowBase + (lane & 15);
    const int xIdx  = (myRow < n_test) ? myRow : (n_test - 1);
    const float xs  = x[xIdx] * sScale;          // pre-scaled test point

    // K-offset for this half-wave: lanes 0-15 -> 0, lanes 16-31 -> 8
    const int halfOff = (lane >> 4) << 3;

    // B = all-ones f16 (any layout of all-ones is all-ones)
    v16h bOnes;
#pragma unroll
    for (int i = 0; i < 16; ++i) bOnes[i] = (_Float16)1.0f;

    v8f acc = {};   // 16x16 f32 accumulator: every column holds the row-sum

    const int nChunks = (n_train + CHUNK - 1) / CHUNK;
    async_copy_chunk(smem[0], tr, (n_train < CHUNK) ? n_train : CHUNK, tid);

    for (int c = 0; c < nChunks; ++c) {
        float* cur        = smem[c & 1];
        const int base    = c * CHUNK;
        const int csize   = (n_train - base < CHUNK) ? (n_train - base) : CHUNK;
        const int cpad    = (csize + 31) & ~31;  // round K up to tile size
        const int loaded  = (csize >> 2) << 2;   // floats covered by async b128s

        asm volatile("s_wait_asynccnt 0x0" ::: "memory");
        __syncthreads();

        // Tail fill (rare): real values for [loaded,csize), sentinel for [csize,cpad).
        // Sentinel 1e30 -> d'^2 overflows -> exp2(-inf) = 0 contribution.
        if (loaded < cpad) {
            for (int i = loaded + tid; i < cpad; i += BLOCK)
                cur[i] = (base + i < n_train) ? tr[base + i] : 1e30f;
            __syncthreads();
        }

        // Kick off next chunk into the other buffer (its last reader finished
        // before the barrier above).
        if (c + 1 < nChunks) {
            const int nbase = (c + 1) * CHUNK;
            const int nsz   = (n_train - nbase < CHUNK) ? (n_train - nbase) : CHUNK;
            async_copy_chunk(smem[(c + 1) & 1], tr + nbase, nsz, tid);
        }

        // March K in tiles of 32; each lane evaluates its 16 A-elements.
#pragma unroll 2
        for (int kk = 0; kk < cpad; kk += 32) {
            const float* p = cur + kk + halfOff;        // 16B-aligned
            float4 t0 = *(const float4*)(p +  0);       // K: off+0..3
            float4 t1 = *(const float4*)(p +  4);       // K: off+4..7
            float4 t2 = *(const float4*)(p + 16);       // K: off+16..19
            float4 t3 = *(const float4*)(p + 20);       // K: off+20..23
            const float tv[16] = { t0.x,t0.y,t0.z,t0.w, t1.x,t1.y,t1.z,t1.w,
                                   t2.x,t2.y,t2.z,t2.w, t3.x,t3.y,t3.z,t3.w };
            union { v16h v; v2h h[8]; } a;
#pragma unroll
            for (int j = 0; j < 8; ++j) {
                const float d0 = __builtin_fmaf(negS, tv[2*j+0], xs); // v_fma
                const float d1 = __builtin_fmaf(negS, tv[2*j+1], xs);
                const float e0 = __builtin_amdgcn_exp2f(-(d0 * d0));  // neg folds into v_mul
                const float e1 = __builtin_amdgcn_exp2f(-(d1 * d1));
                const v2fp16 pk = __builtin_amdgcn_cvt_pkrtz(e0, e1); // v_cvt_pk_rtz_f16_f32
                a.h[j] = __builtin_bit_cast(v2h, pk);
            }
            // D(16x16,f32) += A(16x32,f16) x B(32x16,ones): row sums in every column
            acc = __builtin_amdgcn_wmma_f32_16x16x32_f16(
                /*neg_a=*/false, a.v, /*neg_b=*/false, bOnes,
                /*c_mod=*/(short)0, acc, /*reuse_a=*/false, /*reuse_b=*/false);
        }
        __syncthreads();
    }

    // Column N=0: lane 0 holds rows M=0..7, lane 16 holds M=8..15.
    if ((lane & 15) == 0) {
        const int mBase = rowBase + ((lane >> 4) << 3);
#pragma unroll
        for (int v = 0; v < 8; ++v) {
            const int r = mBase + v;
            if (r < n_test) out[r] = outScale * acc[v];
        }
    }
}

extern "C" void kernel_launch(void* const* d_in, const int* in_sizes, int n_in,
                              void* d_out, int out_size, void* d_ws, size_t ws_size,
                              hipStream_t stream)
{
    const float* x  = (const float*)d_in[0];
    const float* tr = (const float*)d_in[1];
    float* out      = (float*)d_out;
    const int n_test  = in_sizes[0];
    const int n_train = in_sizes[1];

    // BANDWIDTH = 0.05 -> var = 0.0025
    const double var    = 0.05 * 0.05;
    const double two_pi = 6.283185307179586;
    const double log2e  = 1.4426950408889634;
    const double coef   = 1.0 / __builtin_sqrt(two_pi * var);   // ~7.97884561
    const double s      = __builtin_sqrt(log2e / (2.0 * var));  // ~16.9864678

    const float sScale   = (float)s;
    const float negS     = (float)(-s);
    const float outScale = (float)(coef / (double)n_train);

    const int grid = (n_test + ROWS_PER_BLOCK - 1) / ROWS_PER_BLOCK;
    kde_wmma_kernel<<<grid, BLOCK, 0, stream>>>(x, tr, out, n_test, n_train,
                                                sScale, negS, outScale);
}
